// GraphAttentionLayer_68118181315031
// MI455X (gfx1250) — matrix operand order repaired and checked by reading, not compiled, not run on hardware
//
#include <hip/hip_runtime.h>
#include <hip/hip_bf16.h>

typedef __attribute__((ext_vector_type(16))) _Float16 v16h;
typedef __attribute__((ext_vector_type(8)))  float    v8f;

#define GAT_N 2048
#define GAT_B 8
#define GAT_F 128
#define GAT_O 64
#define NEG_BIG (-9.0e15f)
#define LSTR 40      // LDS row stride (halves) for 32-wide tiles
#define K2P  4       // n-range split for column stats

// ---------------------------------------------------------------------------
// Kernel 1: Wh = h @ W ; Wh1 = Wh@a1 ; Wh2 = Wh@a2 ; store Wh transposed f16
// Block: 256 threads, 32 rows of h.  Grid: B * (N/32) = 512 blocks.
// ---------------------------------------------------------------------------
__global__ void __launch_bounds__(256)
k1_project(const float* __restrict__ h, const float* __restrict__ W,
           const float* __restrict__ a, _Float16* __restrict__ whT16,
           float* __restrict__ wh1, float* __restrict__ wh2)
{
    __shared__ float h_lds[32][GAT_F];
    __shared__ float w_lds[GAT_F][GAT_O];
    __shared__ float a1_lds[GAT_O], a2_lds[GAT_O];
    __shared__ float tile[32][GAT_O + 1];
    __shared__ float p1[32][4], p2[32][4];

    const int tid = threadIdx.x;
    const int blk = blockIdx.x;
    const int b   = blk >> 6;
    const int n0  = (blk & 63) * 32;

    const float* hsrc = h + ((size_t)(b * GAT_N + n0)) * GAT_F;
    for (int i = tid; i < 32 * GAT_F; i += 256) ((float*)h_lds)[i] = hsrc[i];
    for (int i = tid; i < GAT_F * GAT_O; i += 256) ((float*)w_lds)[i] = W[i];
    if (tid < GAT_O) { a1_lds[tid] = a[tid]; a2_lds[tid] = a[GAT_O + tid]; }
    __syncthreads();

    const int c  = tid & 63;
    const int rq = tid >> 6;
    #pragma unroll
    for (int i = 0; i < 8; ++i) {
        const int r = rq + 4 * i;
        float acc = 0.f;
        #pragma unroll 8
        for (int k = 0; k < GAT_F; ++k)
            acc = fmaf(h_lds[r][k], w_lds[k][c], acc);
        tile[r][c] = acc;
        whT16[((size_t)(b * GAT_O + c)) * GAT_N + (n0 + r)] = (_Float16)acc;
    }
    __syncthreads();

    if (tid < 128) {
        const int row = tid & 31, q = tid >> 5;
        float s1 = 0.f, s2 = 0.f;
        #pragma unroll
        for (int cc = q * 16; cc < q * 16 + 16; ++cc) {
            const float v = tile[row][cc];
            s1 = fmaf(v, a1_lds[cc], s1);
            s2 = fmaf(v, a2_lds[cc], s2);
        }
        p1[row][q] = s1; p2[row][q] = s2;
    }
    __syncthreads();
    if (tid < 32) {
        wh1[b * GAT_N + n0 + tid] = p1[tid][0] + p1[tid][1] + p1[tid][2] + p1[tid][3];
        wh2[b * GAT_N + n0 + tid] = p2[tid][0] + p2[tid][1] + p2[tid][2] + p2[tid][3];
    }
}

// ---------------------------------------------------------------------------
// Kernel 2a: partial online-softmax stats per column m over an n-range.
// Grid: (N/256, K2P, B); thread = one column, 512 n's. 2048 waves for the
// cold 134 MB adj read (latency hiding for the HBM-bound pass).
// ---------------------------------------------------------------------------
__global__ void __launch_bounds__(256)
k2_colstats_partial(const int* __restrict__ adj, const float* __restrict__ wh1,
                    const float* __restrict__ wh2,
                    float* __restrict__ pM, float* __restrict__ pS)
{
    __shared__ float wh1_lds[GAT_N / K2P];
    const int tid = threadIdx.x;
    const int p   = blockIdx.y;
    const int b   = blockIdx.z;
    const int m   = blockIdx.x * 256 + tid;
    const int nb  = p * (GAT_N / K2P);

    for (int i = tid; i < GAT_N / K2P; i += 256) wh1_lds[i] = wh1[b * GAT_N + nb + i];
    __syncthreads();

    const float w2 = wh2[b * GAT_N + m];
    const int* col = adj + (size_t)b * GAT_N * GAT_N + (size_t)nb * GAT_N + m;

    float M = NEG_BIG, S = 0.f;
    #pragma unroll 4
    for (int n = 0; n < GAT_N / K2P; ++n) {
        const int av = col[(size_t)n * GAT_N];
        float e = wh1_lds[n] + w2;
        e = (e >= 0.f) ? e : 0.2f * e;
        const float x  = (av > 0) ? e : NEG_BIG;
        const float nM = fmaxf(M, x);
        S = S * __expf(M - nM) + __expf(x - nM);
        M = nM;
    }
    pM[((size_t)p * GAT_B + b) * GAT_N + m] = M;
    pS[((size_t)p * GAT_B + b) * GAT_N + m] = S;
}

// ---------------------------------------------------------------------------
// Kernel 2b: deterministic log-sum-exp merge of the K2P partials.
// ---------------------------------------------------------------------------
__global__ void __launch_bounds__(256)
k2_merge(const float* __restrict__ pM, const float* __restrict__ pS,
         float* __restrict__ colM, float* __restrict__ colS)
{
    const int idx = blockIdx.x * 256 + threadIdx.x;   // over B*N
    float Mp[K2P], Sp[K2P];
    float M = NEG_BIG;
    #pragma unroll
    for (int p = 0; p < K2P; ++p) {
        Mp[p] = pM[(size_t)p * GAT_B * GAT_N + idx];
        Sp[p] = pS[(size_t)p * GAT_B * GAT_N + idx];
        M = fmaxf(M, Mp[p]);
    }
    float S = 0.f;
    #pragma unroll
    for (int p = 0; p < K2P; ++p) S += Sp[p] * __expf(Mp[p] - M);
    colM[idx] = M;
    colS[idx] = S;
}

// ---------------------------------------------------------------------------
// Kernel 3: h_prime = attn @ Wh via v_wmma_f32_16x16x32_f16, fused ELU.
// 256 threads = 8 waves; block computes 64(n) x 64(o); wave = 16n x 32o.
// Wh^T tile AND stats staged with global_load_async_to_lds_b{128,32}
// (ASYNCcnt path, no loadcnt stall in the hot loop).
// ---------------------------------------------------------------------------
__global__ void __launch_bounds__(256)
k3_attn_matmul(const int* __restrict__ adj, const _Float16* __restrict__ whT16,
               const float* __restrict__ wh1, const float* __restrict__ stats,
               float* __restrict__ out)
{
    __shared__ _Float16 attn_lds[64][LSTR];
    __shared__ _Float16 whT_lds[64][LSTR];
    __shared__ float wh1_lds[64];
    __shared__ float stat_lds[3][32];     // [0]=wh2, [1]=colM, [2]=colS

    const int tid  = threadIdx.x;
    const int lane = tid & 31;
    const int wave = tid >> 5;            // 0..7
    const int b    = blockIdx.y;
    const int n0   = blockIdx.x * 64;
    const size_t adj_base = (size_t)b * GAT_N * GAT_N;

    if (tid < 64) wh1_lds[tid] = wh1[b * GAT_N + n0 + tid];

    // ---- per-thread staging addresses (loop-invariant parts) ----
    // stats: async b32, uniform base = stats, per-lane byte offset
    const unsigned stat_voff_base =
        (unsigned)(((tid >> 5) * (GAT_B * GAT_N) + b * GAT_N + lane) * 4);
    const unsigned stat_lds_dst = (unsigned)(size_t)&stat_lds[tid >> 5][lane];
    // whT: async b128, uniform base = whT16 + b*O*N, per-lane byte offset
    const _Float16* whT_base = whT16 + (size_t)b * GAT_O * GAT_N;
    const int o_st  = tid >> 2;           // 0..63
    const int j_st  = (tid & 3) * 8;      // 0,8,16,24 halves
    const unsigned lds_dst =
        (unsigned)(size_t)&whT_lds[o_st][j_st];   // low 32b = LDS offset
    const unsigned voff_base = (unsigned)((o_st * GAT_N + j_st) * 2);

    v8f acc[2] = {};

    for (int m0 = 0; m0 < GAT_N; m0 += 32) {
        __syncthreads();   // previous iteration's fragment reads complete

        // ---- stage stats (waves 0..2, async b32 global->LDS) ----
        if (tid < 96) {
            const unsigned svoff = stat_voff_base + (unsigned)(m0 * 4);
            asm volatile("global_load_async_to_lds_b32 %0, %1, %2"
                         :: "v"(stat_lds_dst), "v"(svoff), "s"(stats)
                         : "memory");
        }

        // ---- stage Wh^T f16 tile (32m x 64o, async b128 global->LDS) ----
        {
            const unsigned voff = voff_base + (unsigned)(m0 * 2);
            asm volatile("global_load_async_to_lds_b128 %0, %1, %2"
                         :: "v"(lds_dst), "v"(voff), "s"(whT_base)
                         : "memory");
        }
        asm volatile("s_wait_asynccnt 0x0" ::: "memory");
        __syncthreads();   // stats + whT visible

        // ---- build 64 x 32 attention tile in f16 ----
        {
            const int i  = tid >> 2;              // n-local 0..63
            const int j0 = (tid & 3) * 8;         // m-local 0,8,16,24
            const int n  = n0 + i;
            const float w1 = wh1_lds[i];
            const int* arow = adj + adj_base + (size_t)n * GAT_N + m0 + j0;
            if (m0 + 32 < GAT_N)
                __builtin_prefetch((const void*)(arow + 32), 0, 3);
            int4 a4[2];
            a4[0] = ((const int4*)arow)[0];
            a4[1] = ((const int4*)arow)[1];
            const int* av = (const int*)a4;
            #pragma unroll
            for (int s = 0; s < 8; ++s) {
                const int j = j0 + s;
                float e = w1 + stat_lds[0][j];
                e = (e >= 0.f) ? e : 0.2f * e;
                const float x  = (av[s] > 0) ? e : NEG_BIG;
                const float at = __expf(x - stat_lds[1][j]) / stat_lds[2][j];
                attn_lds[i][j] = (_Float16)at;
            }
        }
        __syncthreads();   // attention tile visible

        // ---- A fragment: 16x32 f16 (rows = this wave's n slice) ----
        v16h afrag;
        {
            const int r  = (wave & 3) * 16 + (lane & 15);
            const int kb = (lane < 16) ? 0 : 8;
            #pragma unroll
            for (int t = 0; t < 8; ++t) afrag[t]     = attn_lds[r][kb + t];
            #pragma unroll
            for (int t = 0; t < 8; ++t) afrag[8 + t] = attn_lds[r][16 + kb + t];
        }

        // ---- 2 o-tiles per wave: B fragment (32x16) + WMMA ----
        #pragma unroll
        for (int ot = 0; ot < 2; ++ot) {
            const int got = (wave >> 2) * 2 + ot;        // global o-tile 0..3
            v16h bfrag;
            const int orow = got * 16 + (lane & 15);
            const int kb2  = (lane < 16) ? 0 : 8;
            #pragma unroll
            for (int t = 0; t < 16; ++t) bfrag[t] = whT_lds[orow][kb2 + t + ((t < 8) ? 0 : 8)];
            acc[ot] = __builtin_amdgcn_wmma_f32_16x16x32_f16(
                false, afrag, false, bfrag, (short)0, acc[ot], false, false);
        }
    }

    // ---- store with fused ELU; C/D layout: VGPR v -> M=v (+8 upper lanes) ----
    const int rbase = n0 + (wave & 3) * 16 + ((lane < 16) ? 0 : 8);
    const int ocol  = lane & 15;
    #pragma unroll
    for (int ot = 0; ot < 2; ++ot) {
        const int o = ((wave >> 2) * 2 + ot) * 16 + ocol;
        #pragma unroll
        for (int vr = 0; vr < 8; ++vr) {
            const int n = rbase + vr;
            float v = acc[ot][vr];
            v = (v > 0.f) ? v : (__expf(v) - 1.f);
            out[((size_t)(b * GAT_N + n)) * GAT_O + o] = v;
        }
    }
}

// ---------------------------------------------------------------------------
extern "C" void kernel_launch(void* const* d_in, const int* in_sizes, int n_in,
                              void* d_out, int out_size, void* d_ws, size_t ws_size,
                              hipStream_t stream)
{
    (void)in_sizes; (void)n_in; (void)out_size; (void)ws_size;
    const float* h   = (const float*)d_in[0];
    const int*   adj = (const int*)  d_in[1];
    const float* W   = (const float*)d_in[2];
    const float* a   = (const float*)d_in[3];
    float* out = (float*)d_out;

    const size_t BN = (size_t)GAT_B * GAT_N;
    char* ws = (char*)d_ws;
    _Float16* whT16 = (_Float16*)ws;                        // 2 MB
    float* wh1   = (float*)(ws + (size_t)2 * 1024 * 1024);  // 64 KB
    float* stats = wh1 + BN;            // [wh2 | colM | colS], contiguous
    float* wh2   = stats;
    float* colM  = stats + BN;
    float* colS  = stats + 2 * BN;
    float* pM    = stats + 3 * BN;      // K2P partials: 256 KB
    float* pS    = pM + (size_t)K2P * BN;

    k1_project<<<dim3(GAT_B * (GAT_N / 32)), dim3(256), 0, stream>>>(
        h, W, a, whT16, wh1, wh2);
    k2_colstats_partial<<<dim3(GAT_N / 256, K2P, GAT_B), dim3(256), 0, stream>>>(
        adj, wh1, wh2, pM, pS);
    k2_merge<<<dim3((GAT_B * GAT_N) / 256), dim3(256), 0, stream>>>(
        pM, pS, colM, colS);
    k3_attn_matmul<<<dim3(GAT_N / 64, GAT_B), dim3(256), 0, stream>>>(
        adj, whT16, wh1, stats, out);
}
